// GNN_4260607558136
// MI455X (gfx1250) — matrix operand order repaired and checked by reading, not compiled, not run on hardware
//
#include <hip/hip_runtime.h>

// ---------------------------------------------------------------------------
// CDNA5 (gfx1250) GNN SAGEConv, bf16 WMMA (f32 accumulate), wave32.
// 64-row M-tiles (B-fragments reused 4x), bf16 LDS staging, and 32-bit
// pre-scaled offsets in the gather/scatter epilogues so the backend can use
// SGPR-base + scaled 32-bit voffset addressing.
// ---------------------------------------------------------------------------

typedef __bf16 bf16_t;
typedef bf16_t v16bf __attribute__((ext_vector_type(16)));
typedef bf16_t v8bf  __attribute__((ext_vector_type(8)));
typedef bf16_t v4bf  __attribute__((ext_vector_type(4)));
typedef float  v8f   __attribute__((ext_vector_type(8)));

#define WMMA_BF16(A, B, C) \
  __builtin_amdgcn_wmma_f32_16x16x32_bf16(false, (A), false, (B), (short)0, (C), false, false)

// Padded LDS row strides (bf16 elements): 144B / 528B rows are 16B-aligned
// with a 16-row bank period -> A-fragment rows hit distinct banks.
#define EA_STRIDE 72
#define XW_STRIDE 264

// A-fragment (16x32 bf16, M = lane&15): two aligned 16B runs
// (K = h*16..+7 and h*16+8..+15) -> two ds_load_b128.
__device__ __forceinline__ v16bf load_a_frag(const bf16_t* row, int h) {
  const v8bf lo = *(const v8bf*)(row + h * 16);
  const v8bf hi = *(const v8bf*)(row + h * 16 + 8);
  return __builtin_shufflevector(lo, hi, 0, 1, 2, 3, 4, 5, 6, 7,
                                 8, 9, 10, 11, 12, 13, 14, 15);
}

// B-fragment (32x16 bf16): lane n = lane&15 holds W[n][kk+16h .. +15]
// (16 consecutive bf16, 32B aligned at all call sites).
__device__ __forceinline__ v16bf load_b_frag(const bf16_t* __restrict__ p) {
  return *(const v16bf*)p;
}

__device__ __forceinline__ v4bf cvt4(float4 v) {
  v4bf o;
  o[0] = (bf16_t)v.x; o[1] = (bf16_t)v.y; o[2] = (bf16_t)v.z; o[3] = (bf16_t)v.w;
  return o;
}

// ---------------------------------------------------------------------------
// Small utility kernels
// ---------------------------------------------------------------------------
__global__ void zero_f32(float* __restrict__ p, long long n) {
  long long i = (long long)blockIdx.x * blockDim.x + threadIdx.x;
  if (i < n) p[i] = 0.0f;
}

__global__ void deg_count(const int* __restrict__ dst, float* __restrict__ deg, int E) {
  int i = blockIdx.x * blockDim.x + threadIdx.x;
  if (i < E) atomicAdd(&deg[dst[i]], 1.0f);
}

__global__ void deg_invert(float* __restrict__ deg, int n) {
  int i = blockIdx.x * blockDim.x + threadIdx.x;
  if (i < n) deg[i] = 1.0f / fmaxf(deg[i], 1.0f);
}

// One-time fp32 -> bf16 weight conversion (n multiple of 4).
__global__ void f32_to_bf16(const float* __restrict__ in, bf16_t* __restrict__ out,
                            long long n) {
  long long i = 4ll * ((long long)blockIdx.x * blockDim.x + threadIdx.x);
  if (i < n) {
    *(v4bf*)(out + i) = cvt4(*(const float4*)(in + i));
  }
}

// ---------------------------------------------------------------------------
// Edge kernel: per 64-edge tile, e_msg = edge_attr @ We^T + be (WMMA),
// msg = x[src] + e_msg, atomic scatter-add into agg[dst].
// Block = 256 threads = 8 waves; wave w owns columns [32w, 32w+32) and
// iterates 4 M-subtiles, reusing each B-fragment 4x.
// ---------------------------------------------------------------------------
__global__ __launch_bounds__(256) void edge_msg_scatter(
    const float*  __restrict__ x,          // [N,256] fp32
    const float*  __restrict__ edge_attr,  // [E,64]  fp32
    const int*    __restrict__ src,        // [E]
    const int*    __restrict__ dst,        // [E]
    const bf16_t* __restrict__ WeB_l,      // [256,64] bf16
    const float*  __restrict__ be_l,       // [256]
    float*        __restrict__ agg,        // [N,256]
    int E) {
  __shared__ __align__(16) bf16_t sEA[64 * EA_STRIDE];
  __shared__ int sSrcOff[64];  // pre-scaled element offsets: src[e]*256
  __shared__ int sDstOff[64];  // pre-scaled element offsets: dst[e]*256

  const int e0  = blockIdx.x * 64;
  const int tid = threadIdx.x;

  // Stage 64x64 tile (4096 floats = 1024 float4, 4 per thread), cvt once.
  {
    const float4* g = (const float4*)(edge_attr + (size_t)e0 * 64);
#pragma unroll
    for (int j = 0; j < 4; ++j) {
      const int idx = tid + j * 256;
      const int row = idx >> 4;   // 16 float4 per 64-float row
      const int c4  = idx & 15;
      *(v4bf*)(sEA + row * EA_STRIDE + c4 * 4) = cvt4(g[idx]);
    }
  }
  if (tid < 64) {
    sSrcOff[tid] = src[e0 + tid] * 256;
    sDstOff[tid] = dst[e0 + tid] * 256;
  }
  // Prefetch next tile (uniform condition; emits global_prefetch_b8).
  if (e0 + 64 < E) __builtin_prefetch(edge_attr + (size_t)(e0 + 64) * 64, 0, 0);
  __syncthreads();

  const int wave = tid >> 5;
  const int lane = tid & 31;
  const int h    = lane >> 4;
  const int nl   = lane & 15;
  const int n0   = wave * 32;

  v8f c[4][2] = {};
#pragma unroll
  for (int kk = 0; kk < 64; kk += 32) {
    const v16bf b0 = load_b_frag(WeB_l + (size_t)(n0 + nl) * 64 + kk + 16 * h);
    const v16bf b1 = load_b_frag(WeB_l + (size_t)(n0 + 16 + nl) * 64 + kk + 16 * h);
#pragma unroll
    for (int ms = 0; ms < 4; ++ms) {
      const v16bf a = load_a_frag(sEA + (ms * 16 + nl) * EA_STRIDE + kk, h);
      c[ms][0] = WMMA_BF16(a, b0, c[ms][0]);
      c[ms][1] = WMMA_BF16(a, b1, c[ms][1]);
    }
  }

  // Epilogue: bias + gather x[src] + atomic scatter to agg[dst] (fp32).
  // 32-bit element offsets (max 12.8M) -> saddr + scaled voffset addressing.
#pragma unroll
  for (int ms = 0; ms < 4; ++ms) {
#pragma unroll
    for (int t = 0; t < 2; ++t) {
      const int n = n0 + 16 * t + nl;
      const float bias = be_l[n];
#pragma unroll
      for (int r = 0; r < 8; ++r) {
        const int m  = ms * 16 + 8 * h + r;  // lanes 16-31 hold rows 8..15
        const int go = sSrcOff[m] + n;
        const int ao = sDstOff[m] + n;
        const float val = c[ms][t][r] + bias + x[go];
        atomicAdd(&agg[ao], val);
      }
    }
  }
}

// ---------------------------------------------------------------------------
// Node kernel: x_out = relu((agg*inv_deg) @ Wl^T + x_in @ Wr^T + bl).
// Block = 256 threads = 8 waves handles a 64-node tile (zero-padded tail);
// both GEMMs accumulate into the same C fragments; each B-fragment reused 4x.
// ---------------------------------------------------------------------------
__global__ __launch_bounds__(256) void node_update(
    const float*  __restrict__ x_in,     // [N,256] fp32
    const float*  __restrict__ agg,      // [N,256] fp32
    const float*  __restrict__ inv_deg,  // [N]
    const bf16_t* __restrict__ WlB_l,    // [256,256] bf16
    const bf16_t* __restrict__ WrB_l,    // [256,256] bf16
    const float*  __restrict__ bl_l,     // [256]
    float*        __restrict__ x_out,    // [N,256]
    int N) {
  __shared__ __align__(16) bf16_t sAgg[64 * XW_STRIDE];
  __shared__ __align__(16) bf16_t sX[64 * XW_STRIDE];

  const int i0  = blockIdx.x * 64;
  const int tid = threadIdx.x;

  // Stage 64x256 tiles: scale agg by inv_deg, cvt to bf16 once; zero-pad OOB.
  {
    const float4* gA = (const float4*)(agg + (size_t)i0 * 256);
    const float4* gX = (const float4*)(x_in + (size_t)i0 * 256);
#pragma unroll
    for (int j = 0; j < 16; ++j) {
      const int idx = tid + j * 256;   // float4 index in [0,4096)
      const int row = idx >> 6;        // 64 float4 per 256-float row
      const int c4  = idx & 63;
      v4bf oa = {};
      v4bf ox = {};
      if (i0 + row < N) {
        const float s = inv_deg[i0 + row];
        float4 va = gA[idx];
        va.x *= s; va.y *= s; va.z *= s; va.w *= s;
        oa = cvt4(va);
        ox = cvt4(gX[idx]);
      }
      *(v4bf*)(sAgg + row * XW_STRIDE + c4 * 4) = oa;
      *(v4bf*)(sX   + row * XW_STRIDE + c4 * 4) = ox;
    }
  }
  __syncthreads();

  const int wave = tid >> 5;
  const int lane = tid & 31;
  const int h    = lane >> 4;
  const int nl   = lane & 15;
  const int n0   = wave * 32;

  v8f c[4][2] = {};
#pragma unroll 1
  for (int kk = 0; kk < 256; kk += 32) {
    const v16bf bL0 = load_b_frag(WlB_l + (size_t)(n0 + nl) * 256 + kk + 16 * h);
    const v16bf bR0 = load_b_frag(WrB_l + (size_t)(n0 + nl) * 256 + kk + 16 * h);
    const v16bf bL1 = load_b_frag(WlB_l + (size_t)(n0 + 16 + nl) * 256 + kk + 16 * h);
    const v16bf bR1 = load_b_frag(WrB_l + (size_t)(n0 + 16 + nl) * 256 + kk + 16 * h);
#pragma unroll
    for (int ms = 0; ms < 4; ++ms) {
      const v16bf aA = load_a_frag(sAgg + (ms * 16 + nl) * XW_STRIDE + kk, h);
      const v16bf aX = load_a_frag(sX + (ms * 16 + nl) * XW_STRIDE + kk, h);
      c[ms][0] = WMMA_BF16(aA, bL0, c[ms][0]);
      c[ms][0] = WMMA_BF16(aX, bR0, c[ms][0]);
      c[ms][1] = WMMA_BF16(aA, bL1, c[ms][1]);
      c[ms][1] = WMMA_BF16(aX, bR1, c[ms][1]);
    }
  }

  // Epilogue: bias + ReLU, guarded stores, 32-bit store offsets.
#pragma unroll
  for (int ms = 0; ms < 4; ++ms) {
#pragma unroll
    for (int t = 0; t < 2; ++t) {
      const int n = n0 + 16 * t + nl;
      const float bias = bl_l[n];
#pragma unroll
      for (int r = 0; r < 8; ++r) {
        const int m = ms * 16 + 8 * h + r;
        if (i0 + m < N) {
          const int ro = (i0 + m) * 256 + n;
          x_out[ro] = fmaxf(c[ms][t][r] + bias, 0.0f);
        }
      }
    }
  }
}

// ---------------------------------------------------------------------------
// Host-side launch
// ---------------------------------------------------------------------------
extern "C" void kernel_launch(void* const* d_in, const int* in_sizes, int n_in,
                              void* d_out, int out_size, void* d_ws, size_t ws_size,
                              hipStream_t stream) {
  (void)in_sizes; (void)n_in; (void)out_size; (void)ws_size;

  const int N = 50000, E = 400000, D = 256, ED = 64, L = 3;

  const float* x0        = (const float*)d_in[0];
  const float* edge_attr = (const float*)d_in[1];
  const int*   eidx      = (const int*)d_in[2];
  const int*   src       = eidx;
  const int*   dst       = eidx + E;
  const float* Wl        = (const float*)d_in[3];
  const float* bl        = (const float*)d_in[4];
  const float* Wr        = (const float*)d_in[5];
  const float* We        = (const float*)d_in[6];
  const float* be        = (const float*)d_in[7];
  float*       out       = (float*)d_out;

  // Workspace: deg [N] f32 | xA [N*D] f32 | agg [N*D] f32 |
  //            WlB, WrB [3*D*D] bf16 | WeB [3*D*ED] bf16   (~104 MB total)
  char*   ws  = (char*)d_ws;
  float*  deg = (float*)ws;
  float*  xA  = (float*)(ws + (((size_t)N * 4 + 255) & ~(size_t)255));
  float*  agg = xA + (size_t)N * D;
  bf16_t* WlB = (bf16_t*)(agg + (size_t)N * D);
  bf16_t* WrB = WlB + (size_t)L * D * D;
  bf16_t* WeB = WrB + (size_t)L * D * D;

  // One-time bf16 weight conversion (tiny; L2-resident thereafter).
  {
    const long long nWl = (long long)L * D * D;
    const long long nWe = (long long)L * D * ED;
    f32_to_bf16<<<(int)((nWl / 4 + 255) / 256), 256, 0, stream>>>(Wl, WlB, nWl);
    f32_to_bf16<<<(int)((nWl / 4 + 255) / 256), 256, 0, stream>>>(Wr, WrB, nWl);
    f32_to_bf16<<<(int)((nWe / 4 + 255) / 256), 256, 0, stream>>>(We, WeB, nWe);
  }

  // In-degree -> reciprocal (computed once; edge_index fixed across layers).
  zero_f32<<<(N + 255) / 256, 256, 0, stream>>>(deg, N);
  deg_count<<<(E + 255) / 256, 256, 0, stream>>>(dst, deg, E);
  deg_invert<<<(N + 255) / 256, 256, 0, stream>>>(deg, N);

  const long long nd = (long long)N * D;
  const float* xin = x0;
  for (int l = 0; l < L; ++l) {
    float* xout = (l == 1) ? xA : out;  // layer0 -> out, layer1 -> xA, layer2 -> out
    zero_f32<<<(int)((nd + 255) / 256), 256, 0, stream>>>(agg, nd);
    edge_msg_scatter<<<E / 64, 256, 0, stream>>>(
        xin, edge_attr, src, dst, WeB + (size_t)l * D * ED, be + (size_t)l * D, agg, E);
    node_update<<<(N + 63) / 64, 256, 0, stream>>>(
        xin, agg, deg, WlB + (size_t)l * D * D, WrB + (size_t)l * D * D,
        bl + (size_t)l * D, xout, N);
    xin = xout;
  }
}
